// SpatioTemporalFocalModulation_37658273251444
// MI455X (gfx1250) — matrix operand order repaired and checked by reading, not compiled, not run on hardware
//
#include <hip/hip_runtime.h>

typedef __bf16 bf16;
typedef __bf16 v16bf __attribute__((ext_vector_type(16)));
typedef float  v8f   __attribute__((ext_vector_type(8)));

#define TILE_M 128
#define TILE_N 64
#define TILE_K 32
#define LDSA_STRIDE 40   // 32 + pad, keeps 16B alignment (80B rows)
#define LDSB_STRIDE 40

static constexpr int RROWS = 100352;   // B*H*W = 32*56*56
static constexpr int CCH   = 192;

union FragU { v16bf v; uint4 q[2]; };

__device__ __forceinline__ float gelu_exact(float x) {
    return 0.5f * x * (1.0f + erff(x * 0.70710678118654752f));
}

// bf16 round-to-nearest-even via integer math
__device__ __forceinline__ bf16 f2bf(float f) {
    union { float f; unsigned u; } a; a.f = f;
    unsigned r = a.u + 0x7FFFu + ((a.u >> 16) & 1u);
    union { unsigned short s; bf16 b; } o; o.s = (unsigned short)(r >> 16);
    return o.b;
}
__device__ __forceinline__ float bf2f(bf16 b) {
    union { unsigned short s; bf16 b; } i; i.b = b;
    union { unsigned u; float f; } o; o.u = ((unsigned)i.s) << 16;
    return o.f;
}

// x-row r = ((bt*56+h)*56+w), bt = 8*bi + t  ->  temporal row m = n*8+t, n=(bi*56+h)*56+w
__device__ __forceinline__ int temporal_row(int r) {
    int bt  = r / 3136;
    int rem = r - bt * 3136;
    int bi  = bt >> 3;
    int t   = bt & 7;
    return (bi * 3136 + rem) * 8 + t;
}

struct GArgs {
    const float* A;       // mode0: x ; mode3: q        (f32, [rows][192])
    const float* A2;      // mode3: mod_s (f32)
    const float* A3;      // mode3: mod_t (f32, temporal-row layout)
    const bf16*  Abf;     // mode1/2: bf16 A matrix [rows][192]
    const bf16*  Bp;      // weight pack, [N][192] bf16 (tap-major for conv)
    const float* bias;    // mode0: bf ; mode2/3: bias
    const float* bias2;   // mode0: bft
    float* q_out;         // mode0
    float* ctx_s;         // mode0 (f32, dwconv input)
    float* gate_s;        // mode0
    bf16*  ctx_t_bf;      // mode0 (bf16, conv1 A input)
    float* gate_t;        // mode0 (write) / mode1 (read)
    bf16*  out_bf;        // mode1: gelu(ctx) as bf16 (next layer's A)
    float* out;           // mode2/3: f32 result
    float* out2;          // mode1: ctx_all f32 accumulator
    int ntaps, pad, gate_l, accflag;
};

// MODE 0: projection (f32 A, scatter epilogue)
// MODE 1: temporal conv (bf16 A, shifted taps, gelu+gate epilogue)
// MODE 2: modulation GEMM (bf16 A, +bias)
// MODE 3: final GEMM, A = q*mod_s*mod_t fused on the fly (f32)
template <int MODE>
__global__ __launch_bounds__(256) void fm_gemm(GArgs g) {
    __shared__ bf16 As[TILE_M * LDSA_STRIDE];
    __shared__ bf16 Bs[TILE_N * LDSB_STRIDE];

    const int tid  = threadIdx.x;
    const int lane = tid & 31;
    const int wid  = tid >> 5;
    const int wm   = wid & 3;     // 4 waves along M (32 rows each)
    const int wn   = wid >> 2;    // 2 waves along N (32 cols each)
    const int mBase = blockIdx.y * TILE_M;
    const int nBase = blockIdx.x * TILE_N;

    v8f acc[2][2];
#pragma unroll
    for (int i = 0; i < 2; ++i)
#pragma unroll
        for (int j = 0; j < 2; ++j)
#pragma unroll
            for (int e = 0; e < 8; ++e) acc[i][j][e] = 0.0f;

    const int ntaps = (MODE == 1) ? g.ntaps : 1;
    const int pad   = (MODE == 1) ? g.pad : 0;

    const int a_row = tid >> 1;          // 0..127
    const int a_kb  = (tid & 1) * 16;    // 0 / 16
    const int b_row = tid >> 2;          // 0..63
    const int b_kb  = (tid & 3) * 8;     // 0/8/16/24
    const int h     = lane >> 4;
    const int nLane = lane & 15;

    for (int tap = 0; tap < ntaps; ++tap) {
        const int shift = tap - pad;
        const bf16* Btap = g.Bp + (size_t)((MODE == 1) ? tap : 0) * CCH * CCH;

        for (int ks = 0; ks < CCH; ks += TILE_K) {
            // ---- stage A tile ----
            if (MODE == 3) {
                const int m  = mBase + a_row;
                const int mt = temporal_row(m);
                const float* pq = g.A  + (size_t)m  * CCH + ks + a_kb;
                const float* pm = g.A2 + (size_t)m  * CCH + ks + a_kb;
                const float* pt = g.A3 + (size_t)mt * CCH + ks + a_kb;
                bf16* dst = As + a_row * LDSA_STRIDE + a_kb;
#pragma unroll
                for (int i = 0; i < 16; ++i) dst[i] = f2bf(pq[i] * pm[i] * pt[i]);
            } else if (MODE == 0) {
                const int m = mBase + a_row;
                const float* pa = g.A + (size_t)m * CCH + ks + a_kb;
                bf16* dst = As + a_row * LDSA_STRIDE + a_kb;
#pragma unroll
                for (int i = 0; i < 16; ++i) dst[i] = f2bf(pa[i]);
            } else {
                // bf16 A: straight b128 copies (with t-masking for conv taps)
                const int m = mBase + a_row;
                bool valid = true;
                int srcm = m;
                if (MODE == 1) {
                    int ts = (m & 7) + shift;           // SAME padding over T=8
                    valid = (ts >= 0) && (ts < 8);
                    srcm = m + shift;
                }
                uint4 u0 = {0, 0, 0, 0}, u1 = {0, 0, 0, 0};
                if (valid) {
                    const uint4* pa =
                        (const uint4*)(g.Abf + (size_t)srcm * CCH + ks + a_kb);
                    u0 = pa[0];
                    u1 = pa[1];
                }
                uint4* dst = (uint4*)(As + a_row * LDSA_STRIDE + a_kb);
                dst[0] = u0;
                dst[1] = u1;
            }
            // ---- stage B tile (bf16 [N][K] pack) ----
            {
                const int n = nBase + b_row;
                const bf16* src = Btap + (size_t)n * CCH + ks + b_kb;
                bf16* dst = Bs + b_row * LDSB_STRIDE + b_kb;
                *(uint4*)dst = *(const uint4*)src;
                if (ks + TILE_K < CCH)   // prefetch next K-step of the pack
                    __builtin_prefetch(src + TILE_K, 0, 3);
            }
            __syncthreads();

            // ---- WMMA: 2x2 tiles of 16x16 per wave ----
#pragma unroll
            for (int si = 0; si < 2; ++si) {
                const int am = wm * 32 + si * 16 + nLane;          // M = lane%16
                const char* abase = (const char*)(As + am * LDSA_STRIDE);
                FragU fa;
                fa.q[0] = *(const uint4*)(abase + 16 * h);         // K = 8h + 0..7
                fa.q[1] = *(const uint4*)(abase + 32 + 16 * h);    // K = 16 + 8h + 0..7
#pragma unroll
                for (int sj = 0; sj < 2; ++sj) {
                    const int bn = wn * 32 + sj * 16 + nLane;      // N = lane%16
                    const char* bbase = (const char*)(Bs + bn * LDSB_STRIDE);
                    FragU fb;
                    fb.q[0] = *(const uint4*)(bbase + 16 * h);      // K = 8h + 0..7
                    fb.q[1] = *(const uint4*)(bbase + 32 + 16 * h); // K = 16 + 8h + 0..7
                    acc[si][sj] = __builtin_amdgcn_wmma_f32_16x16x32_bf16(
                        false, fa.v, false, fb.v, (short)0, acc[si][sj], false, false);
                }
            }
            __syncthreads();
        }
    }

    // ---- epilogue ----
#pragma unroll
    for (int si = 0; si < 2; ++si) {
#pragma unroll
        for (int sj = 0; sj < 2; ++sj) {
#pragma unroll
            for (int r = 0; r < 8; ++r) {
                const int row = mBase + wm * 32 + si * 16 + h * 8 + r;  // C/D: M = r + 8h
                const int col = nBase + wn * 32 + sj * 16 + nLane;      // N = lane%16
                const float v = acc[si][sj][r];
                if (MODE == 0) {
                    if (col < 387) {                       // spatial projection
                        const float y = v + g.bias[col];
                        if (col < 192)      g.q_out [(size_t)row * CCH + col]        = y;
                        else if (col < 384) g.ctx_s [(size_t)row * CCH + (col - 192)] = y;
                        else                g.gate_s[(size_t)row * 3   + (col - 384)] = y;
                    } else if (col < 582) {                // temporal projection
                        const int ct = col - 387;
                        const float y = v + g.bias2[ct];
                        const int mt = temporal_row(row);
                        if (ct < 192) g.ctx_t_bf[(size_t)mt * CCH + ct]       = f2bf(y);
                        else          g.gate_t  [(size_t)mt * 3 + (ct - 192)] = y;
                    }
                } else if (MODE == 1) {
                    const float ge = gelu_exact(v);
                    const size_t oi = (size_t)row * CCH + col;
                    g.out_bf[oi] = f2bf(ge);
                    const float gate = g.gate_t[(size_t)row * 3 + g.gate_l];
                    const float prev = g.accflag ? g.out2[oi] : 0.0f;
                    g.out2[oi] = prev + ge * gate;
                } else {  // MODE 2 / 3
                    g.out[(size_t)row * CCH + col] = v + g.bias[col];
                }
            }
        }
    }
}

// ---- weight pre-pack: f32 -> bf16, [N][K] order (transpose where needed) ----
__global__ void prep_kernel(const float* __restrict__ Wf, const float* __restrict__ Wft,
                            const float* __restrict__ tw0, const float* __restrict__ tw1,
                            const float* __restrict__ Wh, const float* __restrict__ Wht,
                            const float* __restrict__ Wp,
                            bf16* packProj, bf16* packT0, bf16* packT1,
                            bf16* packWh, bf16* packWht, bf16* packWp) {
    const int NP = 640 * 192, NT = 192 * 192;
    const int total = NP + 3 * NT + 5 * NT + 3 * NT;
    for (int i = blockIdx.x * blockDim.x + threadIdx.x; i < total;
         i += gridDim.x * blockDim.x) {
        int j = i;
        if (j < NP) {                       // [Wf | Wft] cols as rows, zero-padded to 640
            const int n = j / 192, k = j % 192;
            float v = 0.0f;
            if (n < 387)      v = Wf [k * 387 + n];
            else if (n < 582) v = Wft[k * 195 + (n - 387)];
            packProj[j] = f2bf(v);
        } else if ((j -= NP) < 3 * NT) {    // tw0[o][c][tap] -> pack[tap][o][c]
            const int tap = j / NT, oc = j % NT, o = oc / 192, c = oc % 192;
            packT0[j] = f2bf(tw0[(o * 192 + c) * 3 + tap]);
        } else if ((j -= 3 * NT) < 5 * NT) {
            const int tap = j / NT, oc = j % NT, o = oc / 192, c = oc % 192;
            packT1[j] = f2bf(tw1[(o * 192 + c) * 5 + tap]);
        } else {
            j -= 5 * NT;
            const int which = j / NT, oc = j % NT, o = oc / 192, c = oc % 192;
            if (which == 0)      packWh [oc] = f2bf(Wh [o * 192 + c]);   // B^T[o][c]=Wh[o][c]
            else if (which == 1) packWht[oc] = f2bf(Wht[o * 192 + c]);
            else                 packWp [oc] = f2bf(Wp [c * 192 + o]);   // transpose
        }
    }
}

// ---- depthwise 2D conv + gelu + gated accumulate (row layout [b*h*w][c]) ----
__global__ void dwconv_kernel(const float* __restrict__ in, const float* __restrict__ w,
                              const float* __restrict__ gate, float* __restrict__ ctx_out,
                              float* __restrict__ ctx_all, int K, int l, int accflag) {
    const size_t i = (size_t)blockIdx.x * blockDim.x + threadIdx.x;
    if (i >= (size_t)RROWS * CCH) return;
    const int c = (int)(i % CCH);
    const int row = (int)(i / CCH);
    const int wp = row % 56, hp = (row / 56) % 56, bt = row / 3136;
    const int p = K >> 1;
    float s = 0.0f;
    for (int kh = 0; kh < K; ++kh) {
        const int hh = hp + kh - p;
        if (hh < 0 || hh >= 56) continue;
        for (int kw = 0; kw < K; ++kw) {
            const int ww = wp + kw - p;
            if (ww < 0 || ww >= 56) continue;
            s += w[(c * K + kh) * K + kw] *
                 in[(size_t)((bt * 56 + hh) * 56 + ww) * CCH + c];
        }
    }
    const float ge = gelu_exact(s);
    ctx_out[i] = ge;
    const float gv = gate[(size_t)row * 3 + l];
    ctx_all[i] = (accflag ? ctx_all[i] : 0.0f) + ge * gv;
}

// ---- temporal mean over T=8 (bf16 ctx), gelu, gated add; emit bf16 ctx_all ----
__global__ void tmean_kernel(const bf16* __restrict__ ctx, const float* __restrict__ gate_t,
                             const float* __restrict__ ctx_all_in,
                             bf16* __restrict__ ctx_all_out) {
    const size_t i = (size_t)blockIdx.x * blockDim.x + threadIdx.x;
    if (i >= (size_t)12544 * CCH) return;
    const int c = (int)(i % CCH);
    const int n = (int)(i / CCH);
    float s = 0.0f;
    for (int t = 0; t < 8; ++t) s += bf2f(ctx[(size_t)(n * 8 + t) * CCH + c]);
    const float ge = gelu_exact(s * 0.125f);
    for (int t = 0; t < 8; ++t) {
        const size_t m = (size_t)(n * 8 + t);
        ctx_all_out[m * CCH + c] = f2bf(ctx_all_in[m * CCH + c] + ge * gate_t[m * 3 + 2]);
    }
}

// ---- spatial mean over H*W per (bt,c) ----
__global__ void smean_kernel(const float* __restrict__ ctx, float* __restrict__ gmean) {
    const int bt = blockIdx.x, c = threadIdx.x;   // 32 blocks x 192 threads
    float s = 0.0f;
    for (int p = 0; p < 3136; ++p) s += ctx[(size_t)(bt * 3136 + p) * CCH + c];
    gmean[bt * CCH + c] = gelu_exact(s / 3136.0f);
}

// ---- add spatial mean term; emit bf16 ctx_all for the mod GEMM ----
__global__ void sfin_kernel(const float* __restrict__ ctx_all_in,
                            const float* __restrict__ gmean,
                            const float* __restrict__ gate_s,
                            bf16* __restrict__ ctx_all_out) {
    const size_t i = (size_t)blockIdx.x * blockDim.x + threadIdx.x;
    if (i >= (size_t)RROWS * CCH) return;
    const int c = (int)(i % CCH);
    const int row = (int)(i / CCH);
    const int bt = row / 3136;
    ctx_all_out[i] =
        f2bf(ctx_all_in[i] + gmean[bt * CCH + c] * gate_s[(size_t)row * 3 + 2]);
}

extern "C" void kernel_launch(void* const* d_in, const int* in_sizes, int n_in,
                              void* d_out, int out_size, void* d_ws, size_t ws_size,
                              hipStream_t stream) {
    (void)in_sizes; (void)n_in; (void)out_size; (void)ws_size;
    const float* x   = (const float*)d_in[0];
    const float* Wf  = (const float*)d_in[1];
    const float* bf_ = (const float*)d_in[2];
    const float* Wft = (const float*)d_in[3];
    const float* bft = (const float*)d_in[4];
    const float* fw0 = (const float*)d_in[5];
    const float* fw1 = (const float*)d_in[6];
    const float* tw0 = (const float*)d_in[7];
    const float* tw1 = (const float*)d_in[8];
    const float* Wh  = (const float*)d_in[9];
    const float* bh  = (const float*)d_in[10];
    const float* Wht = (const float*)d_in[11];
    const float* bht = (const float*)d_in[12];
    const float* Wp  = (const float*)d_in[13];
    const float* bp  = (const float*)d_in[14];

    char* ws = (char*)d_ws;
    size_t off = 0;
    auto alloc = [&](size_t bytes) -> void* {
        void* p = ws + off;
        off += (bytes + 255) & ~(size_t)255;
        return p;
    };
    const size_t RC = (size_t)RROWS * CCH;
    // f32 buffers
    float* q         = (float*)alloc(RC * 4);
    float* ctx_s0    = (float*)alloc(RC * 4);
    float* ctx_s1    = (float*)alloc(RC * 4);
    float* ctx_all_s = (float*)alloc(RC * 4);   // later reused as mod_s
    float* ctx_all_t = (float*)alloc(RC * 4);   // later reused as mod_t
    float* gates_s   = (float*)alloc((size_t)RROWS * 3 * 4);
    float* gates_t   = (float*)alloc((size_t)RROWS * 3 * 4);
    float* gmean     = (float*)alloc(32 * CCH * 4);
    // bf16 A-side mirrors
    bf16* ctx_t0_bf    = (bf16*)alloc(RC * 2);
    bf16* ctx_t1_bf    = (bf16*)alloc(RC * 2);
    bf16* ctx_all_t_bf = (bf16*)alloc(RC * 2);
    bf16* ctx_all_s_bf = (bf16*)alloc(RC * 2);
    // bf16 weight packs
    bf16* packProj = (bf16*)alloc((size_t)640 * 192 * 2);
    bf16* packT0   = (bf16*)alloc((size_t)3 * 192 * 192 * 2);
    bf16* packT1   = (bf16*)alloc((size_t)5 * 192 * 192 * 2);
    bf16* packWh   = (bf16*)alloc((size_t)192 * 192 * 2);
    bf16* packWht  = (bf16*)alloc((size_t)192 * 192 * 2);
    bf16* packWp   = (bf16*)alloc((size_t)192 * 192 * 2);
    float* mod_s = ctx_all_s;   // f32 buffer dead after sfin emits bf16 mirror
    float* mod_t = ctx_all_t;   // f32 buffer dead after tmean emits bf16 mirror

    const int MB = RROWS / TILE_M;   // 784

    prep_kernel<<<256, 256, 0, stream>>>(Wf, Wft, tw0, tw1, Wh, Wht, Wp,
                                         packProj, packT0, packT1, packWh, packWht, packWp);

    // 1) fused projections: [x@Wf | x@Wft]; temporal ctx emitted directly as bf16
    {
        GArgs a = {};
        a.A = x; a.Bp = packProj; a.bias = bf_; a.bias2 = bft;
        a.q_out = q; a.ctx_s = ctx_s0; a.gate_s = gates_s;
        a.ctx_t_bf = ctx_t0_bf; a.gate_t = gates_t;
        fm_gemm<0><<<dim3(10, MB), 256, 0, stream>>>(a);
    }
    // 2) temporal conv layer 1 (K=3): bf16 A, ctx_all_t = gelu*gate0
    {
        GArgs a = {};
        a.Abf = ctx_t0_bf; a.Bp = packT0; a.out_bf = ctx_t1_bf; a.out2 = ctx_all_t;
        a.gate_t = gates_t; a.ntaps = 3; a.pad = 1; a.gate_l = 0; a.accflag = 0;
        fm_gemm<1><<<dim3(3, MB), 256, 0, stream>>>(a);
    }
    // 3) temporal conv layer 2 (K=5): ctx_t1_bf -> ctx_t0_bf, ctx_all_t += gelu*gate1
    {
        GArgs a = {};
        a.Abf = ctx_t1_bf; a.Bp = packT1; a.out_bf = ctx_t0_bf; a.out2 = ctx_all_t;
        a.gate_t = gates_t; a.ntaps = 5; a.pad = 2; a.gate_l = 1; a.accflag = 1;
        fm_gemm<1><<<dim3(3, MB), 256, 0, stream>>>(a);
    }
    // 4) temporal mean term -> bf16 ctx_all mirror
    tmean_kernel<<<(12544 * CCH + 255) / 256, 256, 0, stream>>>(ctx_t0_bf, gates_t,
                                                                ctx_all_t, ctx_all_t_bf);
    // 5) mod_t = ctx_all_t @ Wht^T + bht (bf16 A)
    {
        GArgs a = {};
        a.Abf = ctx_all_t_bf; a.Bp = packWht; a.bias = bht; a.out = mod_t;
        fm_gemm<2><<<dim3(3, MB), 256, 0, stream>>>(a);
    }
    // 6) spatial depthwise convs (f32 VALU path, bandwidth-bound)
    const int NE = (int)((RC + 255) / 256);
    dwconv_kernel<<<NE, 256, 0, stream>>>(ctx_s0, fw0, gates_s, ctx_s1, ctx_all_s, 3, 0, 0);
    dwconv_kernel<<<NE, 256, 0, stream>>>(ctx_s1, fw1, gates_s, ctx_s0, ctx_all_s, 5, 1, 1);
    // 7) spatial mean term -> bf16 ctx_all mirror
    smean_kernel<<<32, 192, 0, stream>>>(ctx_s0, gmean);
    sfin_kernel<<<NE, 256, 0, stream>>>(ctx_all_s, gmean, gates_s, ctx_all_s_bf);
    // 8) mod_s = ctx_all_s @ Wh^T + bh (bf16 A)
    {
        GArgs a = {};
        a.Abf = ctx_all_s_bf; a.Bp = packWh; a.bias = bh; a.out = mod_s;
        fm_gemm<2><<<dim3(3, MB), 256, 0, stream>>>(a);
    }
    // 9) final: out = (q * mod_s * mod_t) @ Wp + bp  (A fused on the fly)
    {
        GArgs a = {};
        a.A = q; a.A2 = mod_s; a.A3 = mod_t; a.Bp = packWp; a.bias = bp;
        a.out = (float*)d_out;
        fm_gemm<3><<<dim3(3, MB), 256, 0, stream>>>(a);
    }
}
